// CondensedLinearFineGrained_13597866459291
// MI455X (gfx1250) — matrix operand order repaired and checked by reading, not compiled, not run on hardware
//
#include <hip/hip_runtime.h>
#include <hip/hip_bf16.h>

#define B_TOK 256
#define IN_F  4096
#define OUT_F 4096
#define K_ELL 256

typedef _Float16 v16h __attribute__((ext_vector_type(16)));
typedef _Float16 v8h  __attribute__((ext_vector_type(8)));
typedef _Float16 v4h  __attribute__((ext_vector_type(4)));
typedef float    v8f  __attribute__((ext_vector_type(8)));

// ---------------------------------------------------------------------------
// Kernel 1: densify the ELL (condensed) weights into a dense f16 matrix
// Wh[OUT_F][IN_F].  S[o,j] = sum_{k: mask[o,k]==j} w[o,k].
// One thread owns one dense row's scatter (sequential over k) so duplicate
// indices accumulate in a fixed order -> deterministic.  f32 accumulation in
// LDS, vectorized f16 writeback.  4 rows/block => 64 KB dynamic LDS.
// ---------------------------------------------------------------------------
__global__ __launch_bounds__(256) void densify_kernel(
    const float* __restrict__ cw, const int* __restrict__ mask,
    _Float16* __restrict__ Wh) {
  extern __shared__ float lds[];            // ROWS * IN_F floats
  const int ROWS = 4;
  const int o_base = blockIdx.x * ROWS;
  const int t = threadIdx.x;

  for (int i = t; i < ROWS * IN_F; i += 256) lds[i] = 0.0f;
  __syncthreads();

  if (t < ROWS) {
    const int o = o_base + t;
    const float* __restrict__ wrow = cw + (size_t)o * K_ELL;
    const int*   __restrict__ mrow = mask + (size_t)o * K_ELL;
    float* acc = lds + t * IN_F;
    for (int k = 0; k < K_ELL; ++k) acc[mrow[k]] += wrow[k];
  }
  __syncthreads();

  for (int i = 4 * t; i < ROWS * IN_F; i += 4 * 256) {
    v4h hv;
    hv[0] = (_Float16)lds[i + 0];
    hv[1] = (_Float16)lds[i + 1];
    hv[2] = (_Float16)lds[i + 2];
    hv[3] = (_Float16)lds[i + 3];
    *(v4h*)(Wh + (size_t)o_base * IN_F + i) = hv;
  }
}

// ---------------------------------------------------------------------------
// Kernel 2: convert f32 activations to f16 (vectorized x4)
// ---------------------------------------------------------------------------
__global__ __launch_bounds__(256) void cvt_f16_kernel(
    const float* __restrict__ x, _Float16* __restrict__ xh, int n4) {
  const int i = blockIdx.x * blockDim.x + threadIdx.x;
  if (i >= n4) return;
  const float4 v = ((const float4*)x)[i];
  v4h h;
  h[0] = (_Float16)v.x;
  h[1] = (_Float16)v.y;
  h[2] = (_Float16)v.z;
  h[3] = (_Float16)v.w;
  ((v4h*)xh)[i] = h;
}

// ---------------------------------------------------------------------------
// Kernel 3: dense GEMM out[b,o] = bias[o] + sum_j Ah[b,j] * Wh[o,j]
// using v_wmma_f32_16x16x32_f16 (wave32).
//
// Register tiling: each wave computes a 32(o) x 32(b) macro-tile as a 2x2
// grid of 16x16 WMMA tiles.  Per 32-wide K step: 2 B fragments + 2 A
// fragments (8 x b128 loads) feed 4 WMMAs -> 2.0 loads/WMMA, and both
// operand streams are reused 2x in registers.  Block = 8 waves
// (2 o-groups x 4 b-groups) = 64(o) x 128(b); grid = (64, 2).
//
// Fragment layouts per CDNA5 ISA 7.12.2:
//   A (16x32 f16): lane(m=l&15, h=l>>4), half i -> K = 8h + (i&7) + 16*(i>=8)
//   B (32x16 f16): lane(n=l&15, h=l>>4), half i -> K = 8h + (i&7) + 16*(i>=8)
//   D (16x16 f32): lane n = l&15, VGPR r -> row m = r + 8h
// ---------------------------------------------------------------------------
__global__ __launch_bounds__(256) void gemm_wmma_kernel(
    const _Float16* __restrict__ Ah,   // [B_TOK][IN_F]
    const _Float16* __restrict__ Wh,   // [OUT_F][IN_F]
    const float* __restrict__ bias,
    float* __restrict__ out) {         // [B_TOK][OUT_F]
  const int lane = threadIdx.x & 31;
  const int wave = threadIdx.x >> 5;   // 0..7
  const int ow = wave & 1;             // 2 o-groups of 32 -> 64 outputs/block
  const int bw = wave >> 1;            // 4 b-groups of 32 -> 128 batch/block
  const int o0 = blockIdx.x * 64 + ow * 32;
  const int b0 = blockIdx.y * 128 + bw * 32;
  const int ml = lane & 15;
  const int h  = lane >> 4;

  const _Float16* __restrict__ arow0 = Ah + (size_t)(b0 + ml) * IN_F + 8 * h;
  const _Float16* __restrict__ arow1 = arow0 + (size_t)16 * IN_F;
  const _Float16* __restrict__ brow0 = Wh + (size_t)(o0 + ml) * IN_F + 8 * h;
  const _Float16* __restrict__ brow1 = brow0 + (size_t)16 * IN_F;

  v8f c00 = {};   // (o-tile 0, b-tile 0)
  v8f c01 = {};   // (o-tile 1, b-tile 0)
  v8f c10 = {};   // (o-tile 0, b-tile 1)
  v8f c11 = {};   // (o-tile 1, b-tile 1)

#pragma unroll 2
  for (int j = 0; j < IN_F; j += 32) {
    // B fragments: halves i<8 at K=8h+i, i>=8 at K=8h+16+(i-8)
    v8h bl0 = *(const v8h*)(brow0 + j);
    v8h bh0 = *(const v8h*)(brow0 + j + 16);
    v16h bf0 = __builtin_shufflevector(bl0, bh0,
        0, 1, 2, 3, 4, 5, 6, 7, 8, 9, 10, 11, 12, 13, 14, 15);
    v8h bl1 = *(const v8h*)(brow1 + j);
    v8h bh1 = *(const v8h*)(brow1 + j + 16);
    v16h bf1 = __builtin_shufflevector(bl1, bh1,
        0, 1, 2, 3, 4, 5, 6, 7, 8, 9, 10, 11, 12, 13, 14, 15);

    // A fragments: halves i<8 at K=8h+i, i>=8 at K=8h+16+(i-8)
    v8h a00 = *(const v8h*)(arow0 + j);
    v8h a01 = *(const v8h*)(arow0 + j + 16);
    v16h af0 = __builtin_shufflevector(a00, a01,
        0, 1, 2, 3, 4, 5, 6, 7, 8, 9, 10, 11, 12, 13, 14, 15);
    v8h a10 = *(const v8h*)(arow1 + j);
    v8h a11 = *(const v8h*)(arow1 + j + 16);
    v16h af1 = __builtin_shufflevector(a10, a11,
        0, 1, 2, 3, 4, 5, 6, 7, 8, 9, 10, 11, 12, 13, 14, 15);

    // 4 WMMAs from 8 loads: every operand fragment used twice.
    c00 = __builtin_amdgcn_wmma_f32_16x16x32_f16(
        false, af0, false, bf0, (short)0, c00, false, false);
    c01 = __builtin_amdgcn_wmma_f32_16x16x32_f16(
        false, af0, false, bf1, (short)0, c01, false, false);
    c10 = __builtin_amdgcn_wmma_f32_16x16x32_f16(
        false, af1, false, bf0, (short)0, c10, false, false);
    c11 = __builtin_amdgcn_wmma_f32_16x16x32_f16(
        false, af1, false, bf1, (short)0, c11, false, false);
  }

  const float bv0 = bias[o0 + ml];
  const float bv1 = bias[o0 + 16 + ml];
#pragma unroll
  for (int r = 0; r < 8; ++r) {
    const int m = r + 8 * h;
    float* row0 = out + (size_t)(b0 + m) * OUT_F;
    float* row1 = out + (size_t)(b0 + 16 + m) * OUT_F;
    row0[o0 + ml]      = c00[r] + bv0;
    row0[o0 + 16 + ml] = c01[r] + bv1;
    row1[o0 + ml]      = c10[r] + bv0;
    row1[o0 + 16 + ml] = c11[r] + bv1;
  }
}

// ---------------------------------------------------------------------------
// Fallback: direct gather (only if the workspace is too small for densify)
// ---------------------------------------------------------------------------
__global__ __launch_bounds__(256) void gather_fallback_kernel(
    const float* __restrict__ in, const float* __restrict__ cw,
    const int* __restrict__ mask, const float* __restrict__ bias,
    float* __restrict__ out) {
  const int o = blockIdx.x * blockDim.x + threadIdx.x;
  const int b = blockIdx.y;
  if (o >= OUT_F) return;
  const float* __restrict__ irow = in + (size_t)b * IN_F;
  const int*   __restrict__ mrow = mask + (size_t)o * K_ELL;
  const float* __restrict__ wrow = cw + (size_t)o * K_ELL;
  float s = 0.0f;
  for (int k = 0; k < K_ELL; ++k) s += wrow[k] * irow[mrow[k]];
  out[(size_t)b * OUT_F + o] = s + bias[o];
}

extern "C" void kernel_launch(void* const* d_in, const int* in_sizes, int n_in,
                              void* d_out, int out_size, void* d_ws,
                              size_t ws_size, hipStream_t stream) {
  (void)in_sizes; (void)n_in; (void)out_size;
  const float* in   = (const float*)d_in[0];   // [B_TOK, IN_F]
  const float* cw   = (const float*)d_in[1];   // [OUT_F, K_ELL]
  const int*   mask = (const int*)d_in[2];     // [OUT_F, K_ELL]
  const float* bias = (const float*)d_in[3];   // [OUT_F]
  float* out = (float*)d_out;                  // [B_TOK, OUT_F]

  const size_t whBytes = (size_t)OUT_F * IN_F * sizeof(_Float16);  // 32 MB
  const size_t ihBytes = (size_t)B_TOK * IN_F * sizeof(_Float16);  //  2 MB

  if (ws_size >= whBytes + ihBytes) {
    _Float16* Wh = (_Float16*)d_ws;
    _Float16* Ih = (_Float16*)((char*)d_ws + whBytes);

    // 1) scatter condensed weights into dense f16 [OUT_F][IN_F]
    densify_kernel<<<OUT_F / 4, 256, 4 * IN_F * sizeof(float), stream>>>(
        cw, mask, Wh);

    // 2) activations f32 -> f16
    const int n4 = (B_TOK * IN_F) / 4;
    cvt_f16_kernel<<<(n4 + 255) / 256, 256, 0, stream>>>(in, Ih, n4);

    // 3) dense WMMA GEMM + bias
    dim3 grid(OUT_F / 64, B_TOK / 128);
    gemm_wmma_kernel<<<grid, 256, 0, stream>>>(Ih, Wh, bias, out);
  } else {
    dim3 grid((OUT_F + 255) / 256, B_TOK);
    gather_fallback_kernel<<<grid, 256, 0, stream>>>(in, cw, mask, bias, out);
  }
}
